// LSTMModel_28595892257022
// MI455X (gfx1250) — matrix operand order repaired and checked by reading, not compiled, not run on hardware
//
#include <hip/hip_runtime.h>
#include <math.h>

// ---------------------------------------------------------------------------
// 2-layer LSTM (B=256, T=512, D_IN=64, H=512) + FC(512->512,relu) + FC(512->1,relu)
//   * bf16 WMMA (v_wmma_f32_16x16x32_bf16) gate GEMM, fp32 accumulate
//   * one launch per (layer, timestep); fused cell update in epilogue
//   * 2-way K-split per 16x16 tile: two waves accumulate interleaved K halves,
//     combine via LDS + s_barrier -> 1024 waves/step (~2 per SIMD32) and half
//     the per-wave serial WMMA chain on the latency-critical recurrence.
//   * weights stay L2-resident (~6 MB bf16 total) across all 512 steps.
// ---------------------------------------------------------------------------

typedef __attribute__((ext_vector_type(16))) __bf16 v16bf;
typedef __attribute__((ext_vector_type(8)))  float  v8f;
typedef unsigned short u16;
typedef unsigned int   u32;

#define BB   256   // batch
#define TT   512   // timesteps
#define DIN  64
#define HH   512   // hidden
#define G4H  (4 * HH)

// ---- helpers ---------------------------------------------------------------

__device__ __forceinline__ u16 f2bf(float x) {
  u32 u = __float_as_uint(x);
  u32 r = (u + 0x7fffu + ((u >> 16) & 1u)) >> 16;   // round-to-nearest-even
  return (u16)r;
}

__device__ __forceinline__ float sigmoidf_fast(float x) {
  return 1.0f / (1.0f + __expf(-x));
}
__device__ __forceinline__ float tanhf_fast(float x) {
  return 2.0f / (1.0f + __expf(-2.0f * x)) - 1.0f;  // 2*sigmoid(2x)-1
}

// A-matrix (16x32 bf16, row-major source): lane m<16 holds row m, K {k..k+7} and
// {k+16..k+23}; lane 16+m holds K {k+8..k+15} and {k+24..k+31}.
// Caller passes p already offset by (lane>>4)*8; halves at +0 and +16 elements.
__device__ __forceinline__ v16bf load_a(const u16* p) {
  union { uint4 q[2]; v16bf v; } t;
  t.q[0] = *reinterpret_cast<const uint4*>(p);
  t.q[1] = *reinterpret_cast<const uint4*>(p + 16);
  return t.v;
}

// B-matrix (32x16 bf16 = W rows): lane n<16 holds column n (== W row n) K {0..7} and
// {16..23}; lane 16+n holds K {8..15} and {24..31} (same order as A).
__device__ __forceinline__ v16bf load_b(const u16* p) {
  union { uint4 q[2]; v16bf v; } t;
  t.q[0] = *reinterpret_cast<const uint4*>(p);
  t.q[1] = *reinterpret_cast<const uint4*>(p + 16);
  return t.v;
}

// ---- conversion / init kernels --------------------------------------------

__global__ void k_f32_to_bf16(const float* __restrict__ src, u16* __restrict__ dst, int n) {
  int i = blockIdx.x * blockDim.x + threadIdx.x;
  if (i < n) dst[i] = f2bf(src[i]);
}

__global__ void k_zero_u32(u32* __restrict__ p, int n) {
  int i = blockIdx.x * blockDim.x + threadIdx.x;
  if (i < n) p[i] = 0u;
}

// ---- fused LSTM timestep ---------------------------------------------------
// gates = [x_t | h] @ [W_ih | W_hh]^T + b, then cell update.
// Block = 256 threads = 8 waves = 4 tiles x 2 K-split waves.
// Grid  = 128 blocks -> 512 tiles (16 batch-tiles x 32 unit-tiles), 1024 waves.
__global__ __launch_bounds__(256) void k_lstm_step(
    const u16*  __restrict__ xA,    // bf16 activations for x-part, row stride lda
    int lda, int Kx,
    const u16*  __restrict__ Wih,   // bf16 [4H, Kx] row-major
    const u16*  __restrict__ hA,    // bf16 h_prev [B, H]
    const u16*  __restrict__ Whh,   // bf16 [4H, H]
    const float* __restrict__ bias, // f32 [4H]  (b_ih + b_hh, PyTorch i,f,g,o)
    float* __restrict__ cState,     // f32 [B, H]
    u16*  __restrict__ hOut)        // bf16 [B, H]
{
  __shared__ float red[4][4][8][32];          // [tileLocal][gate][r][lane] = 16 KB

  const int lane = threadIdx.x & 31;
  const int w    = threadIdx.x >> 5;          // 0..7 wave in block
  const int tl   = w >> 1;                    // 0..3 tile within block
  const int half = w & 1;                     // K-split half
  const int tile = blockIdx.x * 4 + tl;       // 0..511
  const int bt = tile >> 5;                   // 0..15  batch tile
  const int ut = tile & 31;                   // 0..31  unit tile
  const int mA = lane & 15;                   // A row within tile
  const int kA = (lane >> 4) << 3;            // A k sub-offset (0 or 8)
  const int nB = lane & 15;                   // B column (== weight row) in tile
  const int kB = (lane >> 4) << 3;            // B k sub-offset (0 or 8)

  v8f acc[4] = {};                            // i, f, g, o partial sums (fp32)

  // ---- x contribution (interleaved k-blocks: half*32, step 64) ----
  {
    const u16* aRow = xA + (size_t)(bt * 16 + mA) * lda;
    for (int k0 = half * 32; k0 < Kx; k0 += 64) {
      v16bf a = load_a(aRow + k0 + kA);
#pragma unroll
      for (int g = 0; g < 4; ++g) {
        const u16* wr = Wih + (size_t)(g * HH + ut * 16 + nB) * Kx + k0 + kB;
        v16bf b = load_b(wr);
        acc[g] = __builtin_amdgcn_wmma_f32_16x16x32_bf16(
            false, a, false, b, (short)0, acc[g], false, false);
      }
    }
  }

  // ---- h contribution ----
  {
    const u16* hRow = hA + (size_t)(bt * 16 + mA) * HH;
    for (int k0 = half * 32; k0 < HH; k0 += 64) {
      __builtin_prefetch(hRow + k0 + 128, 0, 0);   // global_prefetch_b8
      v16bf a = load_a(hRow + k0 + kA);
#pragma unroll
      for (int g = 0; g < 4; ++g) {
        const u16* wr = Whh + (size_t)(g * HH + ut * 16 + nB) * HH + k0 + kB;
        v16bf b = load_b(wr);
        acc[g] = __builtin_amdgcn_wmma_f32_16x16x32_bf16(
            false, a, false, b, (short)0, acc[g], false, false);
      }
    }
  }

  // ---- combine K-split halves through LDS ----
  if (half) {
#pragma unroll
    for (int g = 0; g < 4; ++g)
#pragma unroll
      for (int r = 0; r < 8; ++r)
        red[tl][g][r][lane] = acc[g][r];
  }
  __syncthreads();
  if (half) return;

#pragma unroll
  for (int g = 0; g < 4; ++g)
#pragma unroll
    for (int r = 0; r < 8; ++r)
      acc[g][r] += red[tl][g][r][lane];

  // ---- fused cell update (even wave only) ----
  // C/D layout: VGPR r, lanes 0-15 -> M=r, N=lane; lanes 16-31 -> M=8+r, N=lane-16
  const int u    = ut * 16 + (lane & 15);
  const int rowB = bt * 16 + ((lane >> 4) << 3);
  const float bi = bias[u];
  const float bf = bias[HH + u];
  const float bg = bias[2 * HH + u];
  const float bo = bias[3 * HH + u];

#pragma unroll
  for (int r = 0; r < 8; ++r) {
    const int idx = (rowB + r) * HH + u;
    float gi = sigmoidf_fast(acc[0][r] + bi);
    float gf = sigmoidf_fast(acc[1][r] + bf);
    float gg = tanhf_fast   (acc[2][r] + bg);
    float go = sigmoidf_fast(acc[3][r] + bo);
    float cv = gf * cState[idx] + gi * gg;
    cState[idx] = cv;
    hOut[idx]   = f2bf(go * tanhf_fast(cv));
  }
}

// ---- FC1: out[256,512] = relu(A_bf16[256,512] @ W_bf16[512,512]^T + b) -----
__global__ __launch_bounds__(256) void k_fc1(
    const u16*  __restrict__ A,
    const u16*  __restrict__ W,
    const float* __restrict__ bias,
    float* __restrict__ out)
{
  const int lane = threadIdx.x & 31;
  const int wave = blockIdx.x * (blockDim.x >> 5) + (threadIdx.x >> 5);
  const int bt = wave >> 5;
  const int ut = wave & 31;
  const int mA = lane & 15;
  const int kA = (lane >> 4) << 3;
  const int nB = lane & 15;
  const int kB = (lane >> 4) << 3;

  v8f acc = {};
  const u16* aRow = A + (size_t)(bt * 16 + mA) * HH;
  for (int k0 = 0; k0 < HH; k0 += 32) {
    v16bf a = load_a(aRow + k0 + kA);
    const u16* wr = W + (size_t)(ut * 16 + nB) * HH + k0 + kB;
    v16bf b = load_b(wr);
    acc = __builtin_amdgcn_wmma_f32_16x16x32_bf16(
        false, a, false, b, (short)0, acc, false, false);
  }

  const int u    = ut * 16 + (lane & 15);
  const int rowB = bt * 16 + ((lane >> 4) << 3);
  const float bv = bias[u];
#pragma unroll
  for (int r = 0; r < 8; ++r)
    out[(rowB + r) * HH + u] = fmaxf(acc[r] + bv, 0.0f);
}

// ---- FC2: out[b] = relu(dot(fc1[b,:], w) + b0), one wave per batch row -----
__global__ __launch_bounds__(256) void k_fc2(
    const float* __restrict__ fc1o,
    const float* __restrict__ w,
    const float* __restrict__ b,
    float* __restrict__ out)
{
  const int lane = threadIdx.x & 31;
  const int row  = blockIdx.x * (blockDim.x >> 5) + (threadIdx.x >> 5);
  if (row >= BB) return;
  float s = 0.0f;
  for (int k = lane; k < HH; k += 32)
    s += fc1o[row * HH + k] * w[k];
  for (int o = 16; o; o >>= 1)
    s += __shfl_down(s, o, 32);
  if (lane == 0) out[row] = fmaxf(s + b[0], 0.0f);
}

// ---------------------------------------------------------------------------

extern "C" void kernel_launch(void* const* d_in, const int* in_sizes, int n_in,
                              void* d_out, int out_size, void* d_ws, size_t ws_size,
                              hipStream_t stream) {
  (void)in_sizes; (void)n_in; (void)out_size; (void)ws_size;

  const float* x     = (const float*)d_in[0];
  const float* Wih0  = (const float*)d_in[1];
  const float* Whh0  = (const float*)d_in[2];
  const float* b0    = (const float*)d_in[3];
  const float* Wih1  = (const float*)d_in[4];
  const float* Whh1  = (const float*)d_in[5];
  const float* b1    = (const float*)d_in[6];
  const float* fc1_w = (const float*)d_in[7];
  const float* fc1_b = (const float*)d_in[8];
  const float* fc_w  = (const float*)d_in[9];
  const float* fc_b  = (const float*)d_in[10];
  float* out = (float*)d_out;

  // bump allocator on workspace (256B aligned)
  size_t off = 0;
  auto alloc = [&](size_t bytes) -> char* {
    size_t a = (off + 255) & ~(size_t)255;
    off = a + bytes;
    return (char*)d_ws + a;
  };

  u16*   x_bf    = (u16*)  alloc((size_t)BB * TT * DIN * 2);   // 16.8 MB
  u16*   Wih0_bf = (u16*)  alloc((size_t)G4H * DIN * 2);
  u16*   Whh0_bf = (u16*)  alloc((size_t)G4H * HH * 2);
  u16*   Wih1_bf = (u16*)  alloc((size_t)G4H * HH * 2);
  u16*   Whh1_bf = (u16*)  alloc((size_t)G4H * HH * 2);
  u16*   fc1w_bf = (u16*)  alloc((size_t)HH * HH * 2);
  u16*   ys0     = (u16*)  alloc((size_t)TT * BB * HH * 2);    // 134 MB, layer-0 h stream
  u16*   h1buf   = (u16*)  alloc((size_t)2 * BB * HH * 2);     // ping-pong layer-1 h
  u16*   h0zero  = (u16*)  alloc((size_t)BB * HH * 2);
  float* c0      = (float*)alloc((size_t)BB * HH * 4);
  float* c1      = (float*)alloc((size_t)BB * HH * 4);
  float* fc1o    = (float*)alloc((size_t)BB * HH * 4);

  auto cvt = [&](const float* s, u16* d, int n) {
    k_f32_to_bf16<<<(n + 255) / 256, 256, 0, stream>>>(s, d, n);
  };
  cvt(x,     x_bf,    BB * TT * DIN);
  cvt(Wih0,  Wih0_bf, G4H * DIN);
  cvt(Whh0,  Whh0_bf, G4H * HH);
  cvt(Wih1,  Wih1_bf, G4H * HH);
  cvt(Whh1,  Whh1_bf, G4H * HH);
  cvt(fc1_w, fc1w_bf, HH * HH);

  // zero initial states (workspace is poisoned; must re-init every launch)
  k_zero_u32<<<(BB * HH + 255) / 256, 256, 0, stream>>>((u32*)c0, BB * HH);
  k_zero_u32<<<(BB * HH + 255) / 256, 256, 0, stream>>>((u32*)c1, BB * HH);
  k_zero_u32<<<(BB * HH / 2 + 255) / 256, 256, 0, stream>>>((u32*)h0zero, BB * HH / 2);

  const dim3 stepGrid(128), blk(256);  // 1024 waves: 512 tiles x 2 K-split waves

  // ---- layer 0: x[b,t,:] (row stride T*DIN), h streamed into ys0 ----
  for (int t = 0; t < TT; ++t) {
    const u16* xA = x_bf + (size_t)t * DIN;
    const u16* hA = (t == 0) ? h0zero : ys0 + (size_t)(t - 1) * BB * HH;
    u16* hO = ys0 + (size_t)t * BB * HH;
    k_lstm_step<<<stepGrid, blk, 0, stream>>>(xA, TT * DIN, DIN,
                                              Wih0_bf, hA, Whh0_bf, b0, c0, hO);
  }

  // ---- layer 1: x = ys0[t] (row stride H), ping-pong h ----
  for (int t = 0; t < TT; ++t) {
    const u16* xA = ys0 + (size_t)t * BB * HH;
    const u16* hA = (t == 0) ? h0zero : h1buf + (size_t)((t - 1) & 1) * BB * HH;
    u16* hO = h1buf + (size_t)(t & 1) * BB * HH;
    k_lstm_step<<<stepGrid, blk, 0, stream>>>(xA, HH, HH,
                                              Wih1_bf, hA, Whh1_bf, b1, c1, hO);
  }

  // hT (bf16) = layer-1 h at t = T-1
  const u16* hT_bf = h1buf + (size_t)((TT - 1) & 1) * BB * HH;

  // ---- FC head ----
  k_fc1<<<dim3(64), blk, 0, stream>>>(hT_bf, fc1w_bf, fc1_b, fc1o);
  k_fc2<<<dim3(32), blk, 0, stream>>>(fc1o, fc_w, fc_b, out);
}
